// Discriminator_16260746183107
// MI455X (gfx1250) — matrix operand order repaired and checked by reading, not compiled, not run on hardware
//
#include <hip/hip_runtime.h>

typedef __attribute__((ext_vector_type(16))) _Float16 v16h;
typedef __attribute__((ext_vector_type(8)))  float    v8f;

#define IMG    128
#define CMPS   181
#define PTS    128
#define NIMG   32
#define MTOT   (127 * 181)      // 22987 interpolated points per image
#define SPLIT  8                // m-chunks per image
#define CHUNK  2880             // SPLIT*CHUNK >= MTOT, multiple of 32
#define KSTEPS (CHUNK / 32)     // 90 K-steps of depth 32

// ---------------------------------------------------------------- init
__global__ void k_init(float* __restrict__ draft, float* __restrict__ out) {
  int i = blockIdx.x * blockDim.x + threadIdx.x;
  if (i < NIMG * IMG * IMG) draft[i] = 0.0f;
  if (i == 0) out[0] = 0.0f;
}

// ---------------------------------------------------------------- distance
__global__ void k_dist(const float* __restrict__ points, float* __restrict__ out) {
  __shared__ float red[256];
  const float X0v = (float)(65024.0 / 6.0);      // mean(2*d^2), d=-127..127
  const float C0 = sqrtf(X0v);
  const float C1 = 0.5f / C0;
  const float C2 = -1.0f / (8.0f * X0v * C0);    // -X0^-1.5 / 8
  const float C3 =  1.0f / (16.0f * X0v * X0v * C0); // X0^-2.5 / 16
  float acc = 0.0f;
  for (int i = threadIdx.x; i < NIMG * (PTS - 1); i += 256) {
    int n = i / (PTS - 1);
    int p = i - n * (PTS - 1);
    const float* a = points + (n * PTS + p) * 3;
    float dx = a[3] - a[0];
    float dy = a[4] - a[1];
    float x  = dx * dx + dy * dy;
    float d  = x - X0v;
    acc += C0 + d * (C1 + d * (C2 + d * C3));
  }
  red[threadIdx.x] = acc;
  __syncthreads();
  for (int s = 128; s > 0; s >>= 1) {
    if (threadIdx.x < s) red[threadIdx.x] += red[threadIdx.x + s];
    __syncthreads();
  }
  if (threadIdx.x == 0) atomicAdd(out, red[0] * (1.0f / 32.0f));
}

// ---------------------------------------------------------------- main GEMM
// draft[n,s,t] += sum_m exp(-(x_m-s)^2/(2w)) * exp(-(y_m-t)^2/(2w))
// One block (8 waves) = one (image, m-chunk). Wave w owns the 16-row strip
// s in [16w,16w+16); accumulates 8 f32 16x16 tiles across t. f16 WMMA K=32.
__global__ __launch_bounds__(256) void k_draft(const float* __restrict__ points,
                                               float* __restrict__ draft) {
  __shared__ __align__(32) _Float16 Bb[8][32][16]; // B fragments, per lane, k = 16h + e
  __shared__ float lx[32], ly[32], liw[32];        // staged line points

  const int tid  = threadIdx.x;
  const int lane = tid & 31;
  const int wave = tid >> 5;
  const int n    = blockIdx.y;
  const int m0   = blockIdx.x * CHUNK;
  const float* pt = points + n * PTS * 3;

  const int   koff = lane & 16;                 // 16 for upper half-wave
  const float srow = (float)(wave * 16 + (lane & 15)); // A row (s)
  const float tcol = (float)(wave * 16 + (lane & 15)); // staged B col (t)

  const v8f vzero = {};
  v8f acc[8];
#pragma unroll
  for (int j = 0; j < 8; ++j) acc[j] = vzero;

  for (int it = 0; it < KSTEPS; ++it) {
    const int mb = m0 + it * 32;

    // stage interpolated (x, y, 1/(2w)) for K = 0..31
    if (tid < 32) {
      int m = mb + tid;
      float xv = 1e30f, yv = 1e30f, iw = 1.0f;   // OOB -> g = exp(-inf) = 0
      if (m < MTOT) {
        int p = m / CMPS;
        int k = m - p * CMPS;
        float t = (float)k * (1.0f / (float)CMPS);
        float u = 1.0f - t;
        const float* a = pt + p * 3;
        xv = u * a[0] + t * a[3];
        yv = u * a[1] + t * a[4];
        float wv = u * a[2] + t * a[5];
        iw = 0.5f / wv;
      }
      lx[tid] = xv; ly[tid] = yv; liw[tid] = iw;
    }
    __syncthreads();

    // A fragment (16x32 f16): lanes<16 K=0-15, lanes>=16 K=16-31
    v16h afrag;
#pragma unroll
    for (int i = 0; i < 16; ++i) {
      int K = i + koff;
      float d = lx[K] - srow;
      afrag[i] = (_Float16)__expf(-d * d * liw[K]);
    }

    // B fragment for tile 'wave' (32x16 f16): lanes<16 K=0-15, lanes>=16 K=16-31
    v16h bstage;
#pragma unroll
    for (int e = 0; e < 16; ++e) {
      int K = e + koff;
      float d = ly[K] - tcol;
      bstage[e] = (_Float16)__expf(-d * d * liw[K]);
    }
    *(v16h*)(&Bb[wave][lane][0]) = bstage;
    __syncthreads();

    // Prefetch ALL 8 B fragments into registers (one burst of ds loads,
    // single dscnt wait), then issue the 8 WMMAs back-to-back so the
    // matrix pipe is not stalled by per-tile LDS load-to-use latency.
    v16h bf[8];
#pragma unroll
    for (int j = 0; j < 8; ++j) {
      bf[j] = *(const v16h*)(&Bb[j][lane][0]);
    }
#pragma unroll
    for (int j = 0; j < 8; ++j) {
      acc[j] = __builtin_amdgcn_wmma_f32_16x16x32_f16(
          /*neg_a=*/false, afrag, /*neg_b=*/false, bf[j],
          /*c_mod=*/(short)0, acc[j], /*reuse_a=*/false, /*reuse_b=*/false);
    }
  }

  // scatter partial tiles: D elem i -> (s = 16w + i + 8*(lane>=16), t = 16j + lane%16)
  float* dn = draft + n * IMG * IMG;
  const int t0  = lane & 15;
  const int shi = (lane >> 4) * 8;
#pragma unroll
  for (int j = 0; j < 8; ++j) {
#pragma unroll
    for (int i = 0; i < 8; ++i) {
      int s = wave * 16 + i + shi;
      int t = j * 16 + t0;
      atomicAdd(dn + s * IMG + t, acc[j][i]);
    }
  }
}

// ---------------------------------------------------------------- tanh + BCE
__global__ void k_bce(const float* __restrict__ draft, const float* __restrict__ img,
                      float* __restrict__ out) {
  __shared__ float red[256];
  int i = blockIdx.x * 256 + threadIdx.x;   // grid covers 32*128*128 exactly
  float d    = draft[i];
  float line = tanhf(d);
  float lp   = fmaxf(__logf(line),        -100.0f);
  float l1   = fmaxf(__logf(1.0f - line), -100.0f);
  float im   = img[i];
  red[threadIdx.x] = -(im * lp + (1.0f - im) * l1);
  __syncthreads();
  for (int s = 128; s > 0; s >>= 1) {
    if (threadIdx.x < s) red[threadIdx.x] += red[threadIdx.x + s];
    __syncthreads();
  }
  // error = mean(bce) * IMG^2 = sum(bce) / N
  if (threadIdx.x == 0) atomicAdd(out, red[0] * (1.0f / 32.0f));
}

// ---------------------------------------------------------------- launch
extern "C" void kernel_launch(void* const* d_in, const int* in_sizes, int n_in,
                              void* d_out, int out_size, void* d_ws, size_t ws_size,
                              hipStream_t stream) {
  const float* points = (const float*)d_in[0];  // (32,128,3)
  const float* img    = (const float*)d_in[1];  // (32,128,128)
  float* out   = (float*)d_out;                 // scalar
  float* draft = (float*)d_ws;                  // 32*128*128 f32 = 2 MB

  int tot = NIMG * IMG * IMG;
  k_init<<<(tot + 255) / 256, 256, 0, stream>>>(draft, out);
  k_dist<<<1, 256, 0, stream>>>(points, out);
  dim3 g2(SPLIT, NIMG);
  k_draft<<<g2, 256, 0, stream>>>(points, draft);
  k_bce<<<tot / 256, 256, 0, stream>>>(draft, img, out);
}
